// KNNClassifier_14027363189008
// MI455X (gfx1250) — matrix operand order repaired and checked by reading, not compiled, not run on hardware
//
#include <hip/hip_runtime.h>
#include <hip/hip_bf16.h>

// ---------------- problem constants (from reference) ----------------
#define N_TRAIN   100000
#define DIM       256
#define N_TEST    2048
#define KNN_K     20
#define NUM_CLS   9
#define SPLITS    32                    // train-dimension splits
#define NPAIRS    (N_TRAIN / 32)        // 3125 (exact) 32-row pair-tiles
#define RUNS      (SPLITS * 2)          // sorted runs per test row
#define WPB       4                     // waves per block (M = 64 per block)

typedef __attribute__((ext_vector_type(16))) __bf16 v16bf;
typedef __attribute__((ext_vector_type(8)))  float  v8f;

union BFVec {
    uint4  u[2];   // 2 x 16B
    v16bf  v;      // 16 bf16 = 32B
};

// round-to-nearest-even f32 -> bf16 (as raw u16)
static __device__ __forceinline__ unsigned short f2bf(float f) {
    unsigned u = __float_as_uint(f);
    unsigned r = ((u >> 16) & 1u) + 0x7FFFu;
    return (unsigned short)((u + r) >> 16);
}

// ---------------------------------------------------------------
// Kernel 1: per-row inverse L2 norm + f32 -> bf16 conversion.
// ---------------------------------------------------------------
__global__ void knn_rownorm_convert(const float* __restrict__ src,
                                    unsigned short* __restrict__ dst_bf,
                                    float* __restrict__ inv_norm,
                                    int nrows) {
    const int wave = threadIdx.x >> 5;
    const int lane = threadIdx.x & 31;
    const int row  = blockIdx.x * 8 + wave;
    if (row >= nrows) return;

    const float* p = src + (size_t)row * DIM + lane * 8;
    float4 f0 = *(const float4*)(p);
    float4 f1 = *(const float4*)(p + 4);

    float s = f0.x * f0.x + f0.y * f0.y + f0.z * f0.z + f0.w * f0.w
            + f1.x * f1.x + f1.y * f1.y + f1.z * f1.z + f1.w * f1.w;
#pragma unroll
    for (int m = 16; m >= 1; m >>= 1) s += __shfl_xor(s, m, 32);

    if (lane == 0) inv_norm[row] = rsqrtf(s);

    uint4 o;
    o.x = (unsigned)f2bf(f0.x) | ((unsigned)f2bf(f0.y) << 16);
    o.y = (unsigned)f2bf(f0.z) | ((unsigned)f2bf(f0.w) << 16);
    o.z = (unsigned)f2bf(f1.x) | ((unsigned)f2bf(f1.y) << 16);
    o.w = (unsigned)f2bf(f1.z) | ((unsigned)f2bf(f1.w) << 16);
    *(uint4*)(dst_bf + (size_t)row * DIM + lane * 8) = o;
}

// ---------------------------------------------------------------
// Kernel 2: fused bf16-WMMA GEMM + per-row top-20, B shared through
// LDS across 4 waves, staged by async global->LDS DMA (no VGPR
// staging, tracked by ASYNCcnt).
// grid = (N_TEST/64, SPLITS), block = 128 (4 waves).
// ---------------------------------------------------------------
__global__ void __launch_bounds__(WPB * 32)
knn_main(const unsigned short* __restrict__ xb,     // [N_TEST][DIM] bf16
         const unsigned short* __restrict__ tb,     // [N_TRAIN][DIM] bf16
         const float* __restrict__ xinv,            // [N_TEST]
         const float* __restrict__ tinv,            // [N_TRAIN]
         float* __restrict__ pvals,                 // [N_TEST][RUNS][K]
         int*   __restrict__ pidx) {                // [N_TEST][RUNS][K]
    __shared__ unsigned short bbuf[2][32 * DIM];    // 2 x 16KB B tiles
    __shared__ float simbuf[WPB][16 * 32];          // per-wave C staging
    __shared__ float topv[WPB][2 * 16 * KNN_K];     // per-wave top lists
    __shared__ int   topi[WPB][2 * 16 * KNN_K];

    const int tid   = threadIdx.x;
    const int wave  = tid >> 5;
    const int lane  = tid & 31;
    const int half  = lane >> 4;       // WMMA layout half
    const int sub   = lane & 15;       // M for A / N for B,C / owned row
    const int m0    = blockIdx.x * (WPB * 16) + wave * 16;
    const int split = blockIdx.y;

    // init this wave's top lists (wave-private: no barrier needed)
    for (int i = lane; i < 2 * 16 * KNN_K; i += 32) {
        topv[wave][i] = -INFINITY;
        topi[wave][i] = 0;
    }
    float th = -INFINITY;              // this lane's list threshold

    // Resident A tile: lane (sub,half) holds row M=sub,
    // K chunks [kc*32 + half*8, +8) and [kc*32 + half*8 + 16, +8).
    v16bf a[8];
    {
        const unsigned short* xrow = xb + (size_t)(m0 + sub) * DIM + half * 8;
#pragma unroll
        for (int kc = 0; kc < 8; ++kc) {
            BFVec t;
            t.u[0] = *(const uint4*)(xrow + kc * 32);
            t.u[1] = *(const uint4*)(xrow + kc * 32 + 16);
            a[kc] = t.v;
        }
    }

    // inverse norms of the 8 test rows this lane touches in C tiles
    float xi[8];
#pragma unroll
    for (int r = 0; r < 8; ++r) xi[r] = xinv[m0 + r + half * 8];

    const int p0 = (split * NPAIRS) / SPLITS;
    const int p1 = ((split + 1) * NPAIRS) / SPLITS;

    // ---- async B staging: a 32-row tile is a contiguous 16KB block of
    // tb. Each thread DMAs 8 x 16B (16B lane stride) straight into LDS
    // via global_load_async_to_lds_b128 (ASYNCcnt; no VGPR staging).
    const unsigned long long tb64 = (unsigned long long)tb;
    const unsigned lds_base[2] = {
        (unsigned)(size_t)&bbuf[0][0],
        (unsigned)(size_t)&bbuf[1][0]
    };
    auto stage_tile_async = [&](int pr, int buf) {
        const unsigned lds  = lds_base[buf] + tid * 16;
        const unsigned voff = (unsigned)pr * (32 * DIM * 2) + tid * 16;
#pragma unroll
        for (int c = 0; c < 8; ++c) {
            // INST_OFFSET is added to BOTH the LDS and global addresses.
            asm volatile(
                "global_load_async_to_lds_b128 %0, %1, %2 offset:%3"
                :
                : "v"(lds), "v"(voff), "s"(tb64), "n"(c * 2048)
                : "memory");
        }
    };

    // prologue: stage first tile
    stage_tile_async(p0, 0);
    asm volatile("s_wait_asynccnt 0x0" ::: "memory");
    __syncthreads();

    for (int pr = p0; pr < p1; ++pr) {
        const int cur = (pr - p0) & 1;
        const int n0  = pr * 32;

        // kick off DMA of the next tile; it flies during compute
        if (pr + 1 < p1) stage_tile_async(pr + 1, cur ^ 1);

        // --- compute: 16 WMMAs against LDS-resident B tile ---
        // B layout: lane (sub,half) reads column N=sub (tile rows sub and
        // sub+16), K chunks [kc*32 + half*8, +8) and [kc*32 + half*8 + 16, +8).
        const unsigned short* bl = &bbuf[cur][0] + (size_t)sub * DIM + half * 8;

        BFVec b0[2], b1[2];
        auto ldb = [&](int kc, int s) {
            const uint4* q0 = (const uint4*)(bl + kc * 32);
            const uint4* q1 = (const uint4*)(bl + 16 * DIM + kc * 32);
            b0[s].u[0] = q0[0]; b0[s].u[1] = q0[2];
            b1[s].u[0] = q1[0]; b1[s].u[1] = q1[2];
        };
        ldb(0, 0);

        v8f c0 = {}, c1 = {};
#pragma unroll
        for (int kc = 0; kc < 8; ++kc) {
            const int s = kc & 1;
            if (kc < 7) ldb(kc + 1, s ^ 1);   // LDS double-buffer: overlap
            c0 = __builtin_amdgcn_wmma_f32_16x16x32_bf16(
                     false, a[kc], false, b0[s].v, (short)0, c0, false, false);
            c1 = __builtin_amdgcn_wmma_f32_16x16x32_bf16(
                     false, a[kc], false, b1[s].v, (short)0, c1, false, false);
        }

        // scale to cosine similarity and stage both tiles (wave-private)
        const float ti0 = tinv[n0 + sub];
        const float ti1 = tinv[n0 + 16 + sub];
#pragma unroll
        for (int r = 0; r < 8; ++r) {
            const int row = r + half * 8;  // C layout rows for this lane
            simbuf[wave][row * 32 + sub]      = c0[r] * xi[r] * ti0;
            simbuf[wave][row * 32 + 16 + sub] = c1[r] * xi[r] * ti1;
        }

        // all 32 lanes scan: lane(sub,half) owns row=sub, cols half*16..+15
        {
            const int base = (half * 16 + sub) * KNN_K;
#pragma unroll 1
            for (int n = 0; n < 16; ++n) {
                const int col = half * 16 + n;
                const float v = simbuf[wave][sub * 32 + col];
                if (v > th) {
                    int p = KNN_K - 1;
                    while (p > 0 && topv[wave][base + p - 1] < v) {
                        topv[wave][base + p] = topv[wave][base + p - 1];
                        topi[wave][base + p] = topi[wave][base + p - 1];
                        --p;
                    }
                    topv[wave][base + p] = v;
                    topi[wave][base + p] = n0 + col;   // global train index
                    th = topv[wave][base + KNN_K - 1];
                }
            }
        }

        // ensure this wave's DMA into the other buffer landed, then sync
        asm volatile("s_wait_asynccnt 0x0" ::: "memory");
        __syncthreads();
    }

    // emit this lane's sorted top-20 run
    {
        const int base = (half * 16 + sub) * KNN_K;
        const size_t pb = ((size_t)(m0 + sub) * RUNS + split * 2 + half) * KNN_K;
        for (int j = 0; j < KNN_K; ++j) {
            pvals[pb + j] = topv[wave][base + j];
            pidx[pb + j]  = topi[wave][base + j];
        }
    }
}

// ---------------------------------------------------------------
// Kernel 3: merge RUNS sorted top-20 runs per test row, vote.
// ---------------------------------------------------------------
__global__ void knn_merge(const float* __restrict__ pvals,
                          const int*   __restrict__ pidx,
                          const int*   __restrict__ labels,
                          float* __restrict__ out) {
    const int i = blockIdx.x * blockDim.x + threadIdx.x;
    if (i >= N_TEST) return;

    float tv[KNN_K];
    int   tx[KNN_K];
#pragma unroll
    for (int j = 0; j < KNN_K; ++j) { tv[j] = -INFINITY; tx[j] = 0; }

    const float* rv = pvals + (size_t)i * RUNS * KNN_K;
    const int*   ri = pidx  + (size_t)i * RUNS * KNN_K;

    for (int s = 0; s < RUNS; ++s) {
        for (int j = 0; j < KNN_K; ++j) {
            const float v = rv[s * KNN_K + j];
            if (v <= tv[KNN_K - 1]) break;   // runs are sorted descending
            int p = KNN_K - 1;
            while (p > 0 && tv[p - 1] < v) {
                tv[p] = tv[p - 1];
                tx[p] = tx[p - 1];
                --p;
            }
            tv[p] = v;
            tx[p] = ri[s * KNN_K + j];
        }
    }

    int cnt[NUM_CLS];
#pragma unroll
    for (int c = 0; c < NUM_CLS; ++c) cnt[c] = 0;
    for (int j = 0; j < KNN_K; ++j) cnt[labels[tx[j]]]++;

    int best = 0;  // ties -> smallest class (torch.mode semantics)
    for (int c = 1; c < NUM_CLS; ++c)
        if (cnt[c] > cnt[best]) best = c;

    out[i] = (float)best;
}

// ---------------------------------------------------------------
extern "C" void kernel_launch(void* const* d_in, const int* in_sizes, int n_in,
                              void* d_out, int out_size, void* d_ws, size_t ws_size,
                              hipStream_t stream) {
    const float* trainF = (const float*)d_in[0];   // [N_TRAIN, DIM] f32
    const int*   labels = (const int*)d_in[1];     // [N_TRAIN] i32
    const float* x      = (const float*)d_in[2];   // [N_TEST, DIM] f32
    float*       out    = (float*)d_out;           // [N_TEST] f32
    (void)in_sizes; (void)n_in; (void)out_size; (void)ws_size;

    // carve workspace (256B aligned)
    char* base = (char*)d_ws;
    size_t off = 0;
    auto carve = [&](size_t bytes) -> char* {
        char* p = base + off;
        off = (off + bytes + 255) & ~(size_t)255;
        return p;
    };
    unsigned short* tb   = (unsigned short*)carve((size_t)N_TRAIN * DIM * 2);
    unsigned short* xb   = (unsigned short*)carve((size_t)N_TEST  * DIM * 2);
    float*          tinv = (float*)carve((size_t)N_TRAIN * 4);
    float*          xinv = (float*)carve((size_t)N_TEST * 4);
    float*          pv   = (float*)carve((size_t)N_TEST * RUNS * KNN_K * 4);
    int*            pi   = (int*)carve((size_t)N_TEST * RUNS * KNN_K * 4);

    // 1) norms + bf16 conversion
    knn_rownorm_convert<<<N_TRAIN / 8, 256, 0, stream>>>(trainF, tb, tinv, N_TRAIN);
    knn_rownorm_convert<<<N_TEST / 8, 256, 0, stream>>>(x, xb, xinv, N_TEST);

    // 2) fused WMMA GEMM + per-split top-20 (two runs per split)
    knn_main<<<dim3(N_TEST / (WPB * 16), SPLITS, 1), WPB * 32, 0, stream>>>(
        xb, tb, xinv, tinv, pv, pi);

    // 3) merge + vote
    knn_merge<<<(N_TEST + 255) / 256, 256, 0, stream>>>(pv, pi, labels, out);
}
